// MambaBlockGated_31086973288936
// MI455X (gfx1250) — matrix operand order repaired and checked by reading, not compiled, not run on hardware
//
#include <hip/hip_runtime.h>
#include <hip/hip_bf16.h>

typedef __attribute__((ext_vector_type(16))) _Float16 v16h;
typedef __attribute__((ext_vector_type(8)))  float    v8f;

#define BQ    2
#define LQ    2048
#define DM    768
#define DS    16
#define DC    4
#define MROWS (BQ * LQ)        // 4096 rows for all GEMMs
#define EPSLN 1e-5f

// async GLOBAL->LDS staging (gfx1250 ASYNCcnt path); set to 0 to fall back to
// the proven global_load_b128 + ds_store_b128 staging.
#define USE_ASYNC_LDS 1

__device__ __forceinline__ float sigmoidf_(float v) { return 1.0f / (1.0f + __expf(-v)); }
__device__ __forceinline__ float siluf_(float v)    { return v / (1.0f + __expf(-v)); }

// low 32 bits of a generic pointer to __shared__ == LDS byte address
__device__ __forceinline__ unsigned lds_lo32_(const void* p) {
    return (unsigned)(unsigned long long)p;
}

// ---------------------------------------------------------------- f32 -> f16
__global__ void cvt_f32_f16(const float* __restrict__ src, _Float16* __restrict__ dst, int n) {
    int i = blockIdx.x * blockDim.x + threadIdx.x;
    if (i < n) dst[i] = (_Float16)src[i];
}

// ---------------------------------------------------------------- LayerNorm -> f16
// one 256-thread block per row of 768
__global__ void layernorm_f16(const float* __restrict__ x, const float* __restrict__ w,
                              const float* __restrict__ b, _Float16* __restrict__ out) {
    const int row = blockIdx.x;
    const int tid = threadIdx.x;              // 256
    const float* xr = x + (size_t)row * DM;
    __shared__ float red[8];
    __shared__ float smu, srs;

    float v0 = xr[tid], v1 = xr[tid + 256], v2 = xr[tid + 512];
    float s = v0 + v1 + v2;
    for (int off = 16; off; off >>= 1) s += __shfl_xor(s, off, 32);
    if ((tid & 31) == 0) red[tid >> 5] = s;
    __syncthreads();
    if (tid == 0) {
        float t = 0.f;
        for (int i = 0; i < 8; ++i) t += red[i];
        smu = t * (1.0f / DM);
    }
    __syncthreads();
    const float mu = smu;
    float d0 = v0 - mu, d1 = v1 - mu, d2 = v2 - mu;
    float s2 = d0 * d0 + d1 * d1 + d2 * d2;
    for (int off = 16; off; off >>= 1) s2 += __shfl_xor(s2, off, 32);
    if ((tid & 31) == 0) red[tid >> 5] = s2;
    __syncthreads();
    if (tid == 0) {
        float t = 0.f;
        for (int i = 0; i < 8; ++i) t += red[i];
        srs = rsqrtf(t * (1.0f / DM) + EPSLN);
    }
    __syncthreads();
    const float rs = srs;
    _Float16* o = out + (size_t)row * DM;
    o[tid]       = (_Float16)(d0 * rs * w[tid]       + b[tid]);
    o[tid + 256] = (_Float16)(d1 * rs * w[tid + 256] + b[tid + 256]);
    o[tid + 512] = (_Float16)(d2 * rs * w[tid + 512] + b[tid + 512]);
}

// ---------------------------------------------------------------- WMMA GEMM
// out[M,N] = A[M,K] * W[N,K]^T + bias[N] (+ resid[M,N])
// block = 128 threads = 4 waves; block tile = 128(M) x 16*NT(N); K-step = 32.
// Each wave owns a 32x(16*NT) strip: 2 A-frags x NT B-frags -> 2*NT WMMAs/K-step.
// Double-buffered LDS; staging via GLOBAL_LOAD_ASYNC_TO_LDS_B128 (ASYNCcnt).
// A/B fragments use contiguous k = 16*(lane/16) + s in both operands; C/D per ISA (wave32).
template <int NT>
__global__ void gemm_wmma_f16(const _Float16* __restrict__ A,  // [M,K]
                              const _Float16* __restrict__ W,  // [N,K]
                              const float* __restrict__ bias,  // [N]
                              const float* __restrict__ resid, // [M,N] or nullptr
                              float* __restrict__ out,         // [M,N]
                              int M, int N, int K) {
    const int tid  = threadIdx.x;   // 128
    const int lane = tid & 31;
    const int wave = tid >> 5;      // 0..3
    const int m0 = blockIdx.x * 128;
    const int n0 = blockIdx.y * (16 * NT);

    __shared__ _Float16 As[2][128][40];       // +8 half pad per row
    __shared__ _Float16 Ws[2][16 * NT][40];

    v8f zero = {};
    v8f acc[2][NT];
#pragma unroll
    for (int mi = 0; mi < 2; ++mi)
#pragma unroll
        for (int nt = 0; nt < NT; ++nt) acc[mi][nt] = zero;

    // stage one 32-wide K-slice of A (128 rows) and W (16*NT rows) into buf.
    // one thread per row: 32 halves = 64 B = 4 x b128.
    auto stage = [&](int kb, int buf) {
        {
            const int r = tid;
            const _Float16* gsrc = A + (size_t)(m0 + r) * K + kb;
            _Float16* ldst = &As[buf][r][0];
#if USE_ASYNC_LDS
            unsigned laddr = lds_lo32_(ldst);
            asm volatile(
                "global_load_async_to_lds_b128 %0, %1, off\n\t"
                "global_load_async_to_lds_b128 %0, %1, off offset:16\n\t"
                "global_load_async_to_lds_b128 %0, %1, off offset:32\n\t"
                "global_load_async_to_lds_b128 %0, %1, off offset:48"
                :: "v"(laddr), "v"(gsrc) : "memory");
#else
            const uint4* src = (const uint4*)gsrc;
            uint4 q0 = src[0], q1 = src[1], q2 = src[2], q3 = src[3];
            *(uint4*)&ldst[0]  = q0;
            *(uint4*)&ldst[8]  = q1;
            *(uint4*)&ldst[16] = q2;
            *(uint4*)&ldst[24] = q3;
#endif
        }
        if (tid < NT * 16) {
            const int r = tid;
            const _Float16* gsrc = W + (size_t)(n0 + r) * K + kb;
            _Float16* ldst = &Ws[buf][r][0];
#if USE_ASYNC_LDS
            unsigned laddr = lds_lo32_(ldst);
            asm volatile(
                "global_load_async_to_lds_b128 %0, %1, off\n\t"
                "global_load_async_to_lds_b128 %0, %1, off offset:16\n\t"
                "global_load_async_to_lds_b128 %0, %1, off offset:32\n\t"
                "global_load_async_to_lds_b128 %0, %1, off offset:48"
                :: "v"(laddr), "v"(gsrc) : "memory");
#else
            const uint4* src = (const uint4*)gsrc;
            uint4 q0 = src[0], q1 = src[1], q2 = src[2], q3 = src[3];
            *(uint4*)&ldst[0]  = q0;
            *(uint4*)&ldst[8]  = q1;
            *(uint4*)&ldst[16] = q2;
            *(uint4*)&ldst[24] = q3;
#endif
        }
    };

    stage(0, 0);
    int cur = 0;
    for (int kb = 0; kb < K; kb += 32) {
#if USE_ASYNC_LDS
        asm volatile("s_wait_asynccnt 0x0" ::: "memory");
#endif
        __syncthreads();                       // staging of buf[cur] visible to all
        if (kb + 32 < K) stage(kb + 32, cur ^ 1);   // overlap next tile with math

        // A fragments: lane<16 -> row M=lane, K 0..15;
        //              lane>=16 -> row M=lane-16, K 16..31
        v16h af[2];
        {
            const int k0 = (lane < 16) ? 0 : 16;
            const int k1 = (lane < 16) ? 8 : 24;
#pragma unroll
            for (int mi = 0; mi < 2; ++mi) {
                const int rl = (wave << 5) + (mi << 4) + (lane & 15);
#pragma unroll
                for (int j = 0; j < 8; ++j) af[mi][j]     = As[cur][rl][k0 + j];
#pragma unroll
                for (int j = 0; j < 8; ++j) af[mi][8 + j] = As[cur][rl][k1 + j];
            }
        }
        // B fragments: lane<16 -> N=lane, K=0..15 ; lane>=16 -> N=lane-16, K=16..31
        v16h bf[NT];
        {
            const int kk = (lane < 16) ? 0 : 16;
#pragma unroll
            for (int nt = 0; nt < NT; ++nt) {
                const int nl = (nt << 4) + (lane & 15);
#pragma unroll
                for (int j = 0; j < 16; ++j) bf[nt][j] = Ws[cur][nl][kk + j];
            }
        }
#pragma unroll
        for (int mi = 0; mi < 2; ++mi)
#pragma unroll
            for (int nt = 0; nt < NT; ++nt)
                acc[mi][nt] = __builtin_amdgcn_wmma_f32_16x16x32_f16(
                    false, af[mi], false, bf[nt], (short)0, acc[mi][nt], false, false);
        cur ^= 1;
    }

    // epilogue: C/D layout — VGPR r: M = r (+8 for hi lanes), N = lane&15
#pragma unroll
    for (int mi = 0; mi < 2; ++mi)
#pragma unroll
        for (int nt = 0; nt < NT; ++nt) {
            const int col = n0 + (nt << 4) + (lane & 15);
            const float bv = bias[col];
#pragma unroll
            for (int r = 0; r < 8; ++r) {
                const int row = m0 + (wave << 5) + (mi << 4) + r + ((lane < 16) ? 0 : 8);
                float v = acc[mi][nt][r] + bv;
                if (resid) v += resid[(size_t)row * N + col];
                out[(size_t)row * N + col] = v;
            }
        }
}

// ------------------------------------------------- causal depthwise conv + SiLU
__global__ void conv_silu(const float* __restrict__ xg,   // [B*L,1536]; x_ssm = cols 0..767
                          const float* __restrict__ cw,   // [768,1,4]
                          const float* __restrict__ cb,   // [768]
                          float* __restrict__ xconv,      // [B*L,768] f32
                          _Float16* __restrict__ xconv_h) {
    int idx = blockIdx.x * blockDim.x + threadIdx.x;
    if (idx >= MROWS * DM) return;
    const int d  = idx % DM;
    const int bl = idx / DM;
    const int l  = bl % LQ;
    const int b  = bl / LQ;
    float acc = cb[d];
#pragma unroll
    for (int j = 0; j < DC; ++j) {
        int pos = l - (DC - 1) + j;
        if (pos >= 0)
            acc += cw[d * DC + j] * xg[((size_t)(b * LQ + pos)) * 1536 + d];
    }
    float s = siluf_(acc);
    xconv[idx]   = s;
    xconv_h[idx] = (_Float16)s;
}

// ------------------------------------------------- sequential gated SSM scan
// 1 block per batch, 1 thread per channel d (768 = 24 wave32s).
// h/ckpt/A_disc in registers; per-step B/C/op in LDS; state-mean via
// shfl_xor wave reduction + ds_add_f32.
__global__ void scan_kernel(const float* __restrict__ xconv,   // [B,L,D]
                            const float* __restrict__ BC,      // [B,L,32]
                            const float* __restrict__ op,      // [B,L,6]
                            const float* __restrict__ A_log,   // [D,16]
                            const float* __restrict__ D_param, // [D]
                            const float* __restrict__ gl,      // [6]
                            const float* __restrict__ collapse,// [16]
                            float* __restrict__ y_out,         // [B,L,D]
                            float* __restrict__ states_out) {  // [B,L,16]
    const int b = blockIdx.x;
    const int d = threadIdx.x;      // 0..767
    const int lane = d & 31;

    __shared__ float sB[DS], sC[DS], sOp[8], sAcc[DS];

    float Ad[DS], h[DS], ck[DS], ct[DS];
#pragma unroll
    for (int s = 0; s < DS; ++s) {
        Ad[s] = __expf(-__expf(A_log[d * DS + s]));
        h[s]  = 0.f;
        ck[s] = 0.f;
        ct[s] = collapse[s];
    }
    const float Dp = D_param[d];
    const float g2 = sigmoidf_(gl[2]);   // OP_IDENTITY
    const float g4 = sigmoidf_(gl[4]);   // OP_CANCEL_END
    const float g5 = sigmoidf_(gl[5]);   // OP_STAR_ZERO

    const size_t bL = (size_t)b * LQ;
    for (int t = 0; t < LQ; ++t) {
        const size_t rt = bL + t;
        if (d < 32) {
            float v = BC[rt * 32 + d];
            if (d < 16) sB[d] = v; else sC[d - 16] = v;
        } else if (d < 38) {
            sOp[d - 32] = op[rt * 6 + (d - 32)];
        } else if (d >= 64 && d < 80) {
            sAcc[d - 64] = 0.f;
        }
        __syncthreads();

        const float xc   = xconv[rt * DM + d];
        const float p_cs = sOp[3];
        const float iw   = sOp[2] * g2;
        const float cw_  = sOp[4] * g4;
        const float zw   = sOp[5] * g5;

        float y = Dp * xc;
#pragma unroll
        for (int s = 0; s < DS; ++s) {
            const float hold = h[s];
            float hn = Ad[s] * hold + xc * sB[s];
            ck[s] = p_cs * hold + (1.f - p_cs) * ck[s];
            hn = (1.f - iw)  * hn + iw  * hold;
            hn = (1.f - cw_) * hn + cw_ * ck[s];
            hn = (1.f - zw)  * hn + zw  * ct[s];
            h[s] = hn;
            y += sC[s] * hn;
        }
        y_out[rt * DM + d] = y;

#pragma unroll
        for (int s = 0; s < DS; ++s) {
            float v = h[s];
            v += __shfl_xor(v, 16, 32);
            v += __shfl_xor(v, 8, 32);
            v += __shfl_xor(v, 4, 32);
            v += __shfl_xor(v, 2, 32);
            v += __shfl_xor(v, 1, 32);
            if (lane == 0) atomicAdd(&sAcc[s], v);
        }
        __syncthreads();
        if (d < DS) states_out[rt * DS + d] = sAcc[d] * (1.0f / DM);
        __syncthreads();   // protect LDS before next step's staging
    }
}

// ------------------------------------------------- y * silu(gate) -> f16
__global__ void gate_mul(const float* __restrict__ y, const float* __restrict__ xg,
                         _Float16* __restrict__ z) {
    int idx = blockIdx.x * blockDim.x + threadIdx.x;
    if (idx >= MROWS * DM) return;
    const int d = idx % DM;
    const size_t row = idx / DM;
    const float g = xg[row * 1536 + DM + d];
    z[idx] = (_Float16)(y[idx] * siluf_(g));
}

// =================================================================== launch
extern "C" void kernel_launch(void* const* d_in, const int* in_sizes, int n_in,
                              void* d_out, int out_size, void* d_ws, size_t ws_size,
                              hipStream_t stream) {
    (void)in_sizes; (void)n_in; (void)out_size; (void)ws_size;
    const float* x    = (const float*)d_in[0];   // [B,L,768]
    const float* op   = (const float*)d_in[1];   // [B,L,6]
    const float* nw   = (const float*)d_in[2];
    const float* nb   = (const float*)d_in[3];
    const float* inw  = (const float*)d_in[4];   // [1536,768]
    const float* inb  = (const float*)d_in[5];
    const float* cw   = (const float*)d_in[6];   // [768,1,4]
    const float* cb   = (const float*)d_in[7];
    const float* xpw  = (const float*)d_in[8];   // [32,768]
    const float* xpb  = (const float*)d_in[9];
    const float* opw  = (const float*)d_in[10];  // [768,768]
    const float* opb  = (const float*)d_in[11];
    const float* alog = (const float*)d_in[12];  // [768,16]
    const float* dpar = (const float*)d_in[13];
    const float* gl   = (const float*)d_in[14];
    const float* ctgt = (const float*)d_in[15];

    char* ws = (char*)d_ws;
    size_t off = 0;
    auto alloc = [&](size_t bytes) -> char* {
        char* p = ws + off;
        off += (bytes + 255) & ~(size_t)255;
        return p;
    };
    _Float16* xn_h  = (_Float16*)alloc((size_t)MROWS * DM * 2);
    _Float16* inw_h = (_Float16*)alloc((size_t)1536 * 768 * 2);
    _Float16* opw_h = (_Float16*)alloc((size_t)768 * 768 * 2);
    _Float16* xpw_h = (_Float16*)alloc((size_t)32 * 768 * 2);
    float*    xg    = (float*)   alloc((size_t)MROWS * 1536 * 4);
    float*    xcv   = (float*)   alloc((size_t)MROWS * DM * 4);
    _Float16* xcv_h = (_Float16*)alloc((size_t)MROWS * DM * 2);
    float*    BCm   = (float*)   alloc((size_t)MROWS * 32 * 4);
    float*    yv    = (float*)   alloc((size_t)MROWS * DM * 4);
    _Float16* z_h   = (_Float16*)alloc((size_t)MROWS * DM * 2);

    float* out1   = (float*)d_out;                       // [B,L,768]
    float* states = out1 + (size_t)MROWS * DM;           // [B,L,16]

    // f16 weight copies (cheap; keeps kernel_launch stateless)
    cvt_f32_f16<<<(1536 * 768 + 255) / 256, 256, 0, stream>>>(inw, inw_h, 1536 * 768);
    cvt_f32_f16<<<(768 * 768 + 255) / 256, 256, 0, stream>>>(opw, opw_h, 768 * 768);
    cvt_f32_f16<<<(32 * 768 + 255) / 256, 256, 0, stream>>>(xpw, xpw_h, 32 * 768);

    layernorm_f16<<<MROWS, 256, 0, stream>>>(x, nw, nb, xn_h);

    // in_proj: [4096,768] x [1536,768]^T
    gemm_wmma_f16<4><<<dim3(MROWS / 128, 1536 / 64), 128, 0, stream>>>(
        xn_h, inw_h, inb, nullptr, xg, MROWS, 1536, 768);

    conv_silu<<<(MROWS * DM + 255) / 256, 256, 0, stream>>>(xg, cw, cb, xcv, xcv_h);

    // x_proj: [4096,768] x [32,768]^T
    gemm_wmma_f16<2><<<dim3(MROWS / 128, 1), 128, 0, stream>>>(
        xcv_h, xpw_h, xpb, nullptr, BCm, MROWS, 32, 768);

    scan_kernel<<<BQ, DM, 0, stream>>>(xcv, BCm, op, alog, dpar, gl, ctgt, yv, states);

    gate_mul<<<(MROWS * DM + 255) / 256, 256, 0, stream>>>(yv, xg, z_h);

    // out_proj + bias + residual, straight into d_out
    gemm_wmma_f16<4><<<dim3(MROWS / 128, 768 / 64), 128, 0, stream>>>(
        z_h, opw_h, opb, x, out1, MROWS, 768, 768);
}
